// RelationalContextEncoder_8650064134431
// MI455X (gfx1250) — matrix operand order repaired and checked by reading, not compiled, not run on hardware
//
#include <hip/hip_runtime.h>
#include <hip/hip_bf16.h>
#include <math.h>

// ---------------- problem constants ----------------
#define NUM_ENT     7
#define DIM_ENT     256
#define REL_HIDDEN  256
#define HIDDEN      1024
#define BATCHN      16384
#define CTX_W       (NUM_ENT * DIM_ENT)        // 1792
#define IN_W        (CTX_W + REL_HIDDEN)       // 2048
#define NPAIR       21

// LDS layout (per 16-row batch tile)
#define A_STRIDE    2056                       // 2048 + 8 pad (row = 4112 B, 16B aligned)
#define STG_CS      16                         // stage col-major: [256][STG_CS], 32B rows

// WMMA vector types (probe-confirmed signatures)
typedef __attribute__((ext_vector_type(16))) __bf16 v16bf;
typedef __attribute__((ext_vector_type(8)))  __bf16 bf16x8;
typedef __attribute__((ext_vector_type(8)))  float  v8f;
typedef __attribute__((ext_vector_type(4)))  float  f32x4;

// combinations(range(7), 2) in lexicographic order
__constant__ int PI_[NPAIR] = {0,0,0,0,0,0,1,1,1,1,1,2,2,2,2,3,3,3,4,4,5};
__constant__ int PJ_[NPAIR] = {1,2,3,4,5,6,2,3,4,5,6,3,4,5,6,4,5,6,5,6,6};

// ---------------------------------------------------------------------------
// Weight repack: f32 row-major [K,N] -> bf16 WMMA B-fragment order.
// Fragment block for (ntile, kstep): 32 lanes x 16 bf16 contiguous (32 B/lane).
// Packing used here (contiguous k, not ISA table order): lanes 0-15 hold K = ks*32+0..15 of col
// (lane&15); lanes 16-31 hold K = ks*32+16..31.
// ---------------------------------------------------------------------------
__global__ void prep_wrel(const float* __restrict__ W, __bf16* __restrict__ F) {
  int idx  = blockIdx.x * 256 + threadIdx.x;   // 16 nt * 16 ks * 32 * 16 = 131072
  int e    = idx & 15;
  int lane = (idx >> 4) & 31;
  int ks   = (idx >> 9) & 15;
  int nt   = idx >> 13;
  int k    = ks * 32 + (lane >> 4) * 16 + e;
  int n    = nt * 16 + (lane & 15);
  F[idx] = (__bf16)W[(size_t)k * REL_HIDDEN + n];
}

__global__ void prep_wfc(const float* __restrict__ W, __bf16* __restrict__ F) {
  int idx  = blockIdx.x * 256 + threadIdx.x;   // 64 nt * 64 ks * 32 * 16 = 2097152
  int e    = idx & 15;
  int lane = (idx >> 4) & 31;
  int ks   = (idx >> 9) & 63;
  int nt   = idx >> 15;
  int k    = ks * 32 + (lane >> 4) * 16 + e;
  int n    = nt * 16 + (lane & 15);
  F[idx] = (__bf16)W[(size_t)k * HIDDEN + n];
}

// Build a v16bf A-fragment from LDS row-major bf16 data.
// A k order used here (contiguous, not ISA table order): lane half h = lane>>4, row = lane&15;
// elements 0..15 -> K = base + h*16 + e (matches B-fragment packing)
__device__ __forceinline__ v16bf load_a_frag(const __bf16* __restrict__ A,
                                             int arow, int colbase, int half) {
  const __bf16* p = A + arow * A_STRIDE + colbase + half * 16;
  bf16x8 lo = *(const bf16x8*)(p);        // ds_load_b128
  bf16x8 hi = *(const bf16x8*)(p + 8);    // ds_load_b128
  return __builtin_shufflevector(lo, hi, 0,1,2,3,4,5,6,7,8,9,10,11,12,13,14,15);
}

// Load a pre-swizzled B-fragment (32 contiguous bytes per lane) from global.
__device__ __forceinline__ v16bf load_b_frag(const __bf16* __restrict__ F,
                                             int blockIdxFrag, int lane) {
  const __bf16* p = F + (((size_t)blockIdxFrag * 32 + lane) << 4);
  bf16x8 lo = *(const bf16x8*)(p);        // global_load_b128
  bf16x8 hi = *(const bf16x8*)(p + 8);    // global_load_b128
  return __builtin_shufflevector(lo, hi, 0,1,2,3,4,5,6,7,8,9,10,11,12,13,14,15);
}

// ---------------------------------------------------------------------------
// Fused kernel: one workgroup (8 wave32) per 16 batch rows.
//  Phase 1: ctx tile f32 -> bf16 into LDS "inpt" buffer cols [0,1792)
//  Phase 2: 21 pair-GEMMs (16x512x256) with WMMA bf16, tanh, staged in LDS,
//           then the torch-faithful permuted window-sum into relacc (no atomics)
//  Phase 3: relacc -> bf16 into inpt cols [1792,2048)
//  Phase 4: fc GEMM (16x2048x1024) with WMMA bf16 + bias + tanh -> d_out f32
// Loop order is K-outer with multiple accumulators so each A-fragment
// (2x ds_load_b128) feeds 2 WMMAs in GEMM1 and 8 WMMAs in GEMM2.
// ---------------------------------------------------------------------------
__global__ __launch_bounds__(256)
void rce_fused(const float*  __restrict__ ctx,
               const __bf16* __restrict__ WrelF,
               const float*  __restrict__ b_rel,
               const __bf16* __restrict__ WfcF,
               const float*  __restrict__ b_fc,
               float*        __restrict__ out) {
  extern __shared__ char smem[];
  __bf16* A      = (__bf16*)smem;                                      // [16][A_STRIDE]
  float*  relacc = (float*)(smem + 16 * A_STRIDE * 2);                 // [16][256]
  __bf16* stage  = (__bf16*)(smem + 16 * A_STRIDE * 2 + 16 * 256 * 4); // [256][STG_CS]

  const int tid  = threadIdx.x;
  const int lane = tid & 31;
  const int wave = tid >> 5;
  const int half = lane >> 4;
  const int arow = lane & 15;     // A-fragment row owned by this lane
  const int ncol = lane & 15;     // C/D-fragment column owned by this lane
  const int row0 = blockIdx.x * 16;

  // ---- Phase 1: ctx tile -> LDS bf16 (float4 loads, 28 iters/thread) ----
  for (int idx = tid; idx < 16 * (CTX_W / 4); idx += 256) {
    int r  = idx / (CTX_W / 4);
    int c4 = (idx % (CTX_W / 4)) * 4;
    f32x4 v = *(const f32x4*)(ctx + (size_t)(row0 + r) * CTX_W + c4);
    __bf16* dst = A + r * A_STRIDE + c4;
    dst[0] = (__bf16)v[0]; dst[1] = (__bf16)v[1];
    dst[2] = (__bf16)v[2]; dst[3] = (__bf16)v[3];
  }
  // zero owned relacc slots (owner thread = tid -> row tid>>4, h = (tid&15)*16..+15)
  {
    int row = tid >> 4, hbase = (tid & 15) * 16;
    for (int hh = 0; hh < 16; ++hh) relacc[row * 256 + hbase + hh] = 0.0f;
  }
  __syncthreads();

  // ---- Phase 2: pair GEMMs + permuted reduction ----
  for (int q = 0; q < NPAIR; ++q) {
    const int pi = PI_[q], pj = PJ_[q];
    const int nt0 = wave * 2;                    // 16 N-tiles over 8 waves
    v8f acc0 = {0.f, 0.f, 0.f, 0.f, 0.f, 0.f, 0.f, 0.f};
    v8f acc1 = {0.f, 0.f, 0.f, 0.f, 0.f, 0.f, 0.f, 0.f};
#pragma unroll
    for (int ks = 0; ks < 16; ++ks) {            // K = 512 in 32-wide steps
      const int colbase = ((ks < 8) ? pi : pj) * DIM_ENT + (ks & 7) * 32;
      v16bf af  = load_a_frag(A, arow, colbase, half);
      v16bf bf0 = load_b_frag(WrelF, (nt0 + 0) * 16 + ks, lane);
      v16bf bf1 = load_b_frag(WrelF, (nt0 + 1) * 16 + ks, lane);
      acc0 = __builtin_amdgcn_wmma_f32_16x16x32_bf16(
          false, af, false, bf0, (short)0, acc0, false, false);
      acc1 = __builtin_amdgcn_wmma_f32_16x16x32_bf16(
          false, af, false, bf1, (short)0, acc1, false, false);
    }
    // epilogue: bias + tanh, one 16B ds_store_b128 per lane per tile
    {
      const float bias0 = b_rel[(nt0 + 0) * 16 + ncol];
      const float bias1 = b_rel[(nt0 + 1) * 16 + ncol];
      bf16x8 p0, p1;
#pragma unroll
      for (int v = 0; v < 8; ++v) {
        p0[v] = (__bf16)tanhf(acc0[v] + bias0);
        p1[v] = (__bf16)tanhf(acc1[v] + bias1);
      }
      // stage col-major: [h'][m]; lane's rows m = half*8 + 0..7 contiguous
      *(bf16x8*)(stage + ((nt0 + 0) * 16 + ncol) * STG_CS + half * 8) = p0;
      *(bf16x8*)(stage + ((nt0 + 1) * 16 + ncol) * STG_CS + half * 8) = p1;
    }
    __syncthreads();
    // torch-faithful sum: rel_out[b,h] = sum_{k in [h*21, h*21+21)} flat[b,k],
    // flat[b, q*256+h'] = stage[h'][b].  Owner-exclusive accumulate.
    {
      const int row = tid >> 4, hbase = (tid & 15) * 16;
      const int qlo = q * 256, qhi = qlo + 256;
      for (int hh = 0; hh < 16; ++hh) {
        const int h  = hbase + hh;
        const int kb = h * 21;
        const int lo = kb > qlo ? kb : qlo;
        const int hi = (kb + 21) < qhi ? (kb + 21) : qhi;
        if (lo < hi) {
          float s = 0.0f;
          for (int k = lo; k < hi; ++k)
            s += (float)stage[(k - qlo) * STG_CS + row];
          relacc[row * 256 + h] += s;
        }
      }
    }
    __syncthreads();
  }

  // ---- Phase 3: rel_out -> bf16 into inpt cols [1792, 2048) ----
  {
    const int row = tid >> 4, hbase = (tid & 15) * 16;
    for (int hh = 0; hh < 16; ++hh)
      A[row * A_STRIDE + CTX_W + hbase + hh] =
          (__bf16)relacc[row * 256 + hbase + hh];
  }
  __syncthreads();

  // ---- Phase 4: fc GEMM 16 x 2048 x 1024 + bias + tanh -> out ----
  {
    v8f acc[8];
#pragma unroll
    for (int t = 0; t < 8; ++t)
      acc[t] = (v8f){0.f, 0.f, 0.f, 0.f, 0.f, 0.f, 0.f, 0.f};
#pragma unroll 4
    for (int ks = 0; ks < 64; ++ks) {            // K = 2048
      v16bf af = load_a_frag(A, arow, ks * 32, half);   // 1 A-frag feeds 8 WMMAs
#pragma unroll
      for (int t = 0; t < 8; ++t) {
        const int nt = wave + t * 8;             // 64 N-tiles over 8 waves
        v16bf bf_ = load_b_frag(WfcF, nt * 64 + ks, lane);
        acc[t] = __builtin_amdgcn_wmma_f32_16x16x32_bf16(
            false, af, false, bf_, (short)0, acc[t], false, false);
      }
    }
#pragma unroll
    for (int t = 0; t < 8; ++t) {
      const int nt = wave + t * 8;
      const float bias = b_fc[nt * 16 + ncol];
#pragma unroll
      for (int v = 0; v < 8; ++v) {
        const int m = v + half * 8;
        out[(size_t)(row0 + m) * HIDDEN + nt * 16 + ncol] = tanhf(acc[t][v] + bias);
      }
    }
  }
}

// ---------------------------------------------------------------------------
extern "C" void kernel_launch(void* const* d_in, const int* in_sizes, int n_in,
                              void* d_out, int out_size, void* d_ws, size_t ws_size,
                              hipStream_t stream) {
  const float* ctx   = (const float*)d_in[0];   // [16384, 1792]
  const float* W_rel = (const float*)d_in[1];   // [512, 256]
  const float* b_rel = (const float*)d_in[2];   // [256]
  const float* W_fc  = (const float*)d_in[3];   // [2048, 1024]
  const float* b_fc  = (const float*)d_in[4];   // [1024]
  float*       out   = (float*)d_out;           // [1, 16384, 1024]

  // ws: bf16 fragment-packed weights (L2-resident, shared by all WGs)
  __bf16* wrelF = (__bf16*)d_ws;                          // 131072 el = 256 KB
  __bf16* wfcF  = (__bf16*)((char*)d_ws + 131072 * 2);    // 2097152 el = 4 MB

  prep_wrel<<<131072 / 256, 256, 0, stream>>>(W_rel, wrelF);
  prep_wfc<<<2097152 / 256, 256, 0, stream>>>(W_fc, wfcF);

  const size_t smem = (size_t)16 * A_STRIDE * 2   // inpt tile bf16   (65792 B)
                    + (size_t)16 * 256 * 4        // relacc f32       (16384 B)
                    + (size_t)256 * STG_CS * 2;   // stage bf16       ( 8192 B)
  rce_fused<<<BATCHN / 16, 256, smem, stream>>>(ctx, wrelF, b_rel, wfcF, b_fc, out);
}
